// MRNNlayer_21861383536866
// MI455X (gfx1250) — matrix operand order repaired and checked by reading, not compiled, not run on hardware
//
#include <hip/hip_runtime.h>
#include <math.h>

typedef __attribute__((ext_vector_type(16))) _Float16 v16h;
typedef __attribute__((ext_vector_type(8)))  _Float16 v8h;
typedef __attribute__((ext_vector_type(8)))  float    v8f;

constexpr int TT = 512;   // time steps
constexpr int BB = 64;    // batch
constexpr int DD = 512;   // D_IN == D_OUT

// ---------------------------------------------------------------------------
// CDNA5 async global->LDS copy (ASYNCcnt path, ISA ch.10 / 15.18):
// each active lane copies 16 bytes MEM[gaddr] -> LDS[lds_off].
// ---------------------------------------------------------------------------
__device__ __forceinline__ void async_copy_b128(unsigned lds_off, const void* gaddr) {
    asm volatile("global_load_async_to_lds_b128 %0, %1, off"
                 :: "v"(lds_off), "v"(gaddr) : "memory");
}
__device__ __forceinline__ void wait_asynccnt0() {
    asm volatile("s_wait_asynccnt 0x0" ::: "memory");
}

// ---------------------------------------------------------------------------
// Fragment loaders following CDNA5 ISA 7.12.2 wave32 layouts.
// A-matrix 16x32 f16: lane L -> row = L&15, hi = L>>4.
//   elements 0..7  = K 8*hi + 0..7 ; elements 8..15 = K 16 + 8*hi + 0..7
// ---------------------------------------------------------------------------
__device__ __forceinline__ v16h load_afrag_f16(const _Float16* __restrict__ base,
                                               int ld, int lane) {
    const int row = lane & 15;
    const int hi  = lane >> 4;
    const _Float16* p = base + row * ld + hi * 8;
    v8h lo = *(const v8h*)(p);
    v8h up = *(const v8h*)(p + 16);
    v16h a;
#pragma unroll
    for (int i = 0; i < 8; ++i) { a[i] = lo[i]; a[i + 8] = up[i]; }
    return a;
}

__device__ __forceinline__ v16h load_afrag_f32(const float* __restrict__ base,
                                               int ld, int lane) {
    const int row = lane & 15;
    const int hi  = lane >> 4;
    const float* p = base + row * ld + hi * 8;
    v16h a;
#pragma unroll
    for (int i = 0; i < 8; ++i) {
        a[i]     = (_Float16)p[i];
        a[i + 8] = (_Float16)p[i + 16];
    }
    return a;
}

// B-matrix 32x16 f16 (weights pre-transposed: WT[n][k], ld = K stride):
// lane L -> col = L&15, hi = L>>4; elements 0..7 = K 8*hi + 0..7 ; elements 8..15 = K 16 + 8*hi + 0..7
__device__ __forceinline__ v16h load_bfrag(const _Float16* __restrict__ wt,
                                           int ld, int lane) {
    const int col = lane & 15;
    const int hi  = lane >> 4;
    const _Float16* p = wt + col * ld + hi * 8;
    v8h lo = *(const v8h*)(p);
    v8h up = *(const v8h*)(p + 16);
    v16h b;
#pragma unroll
    for (int i = 0; i < 8; ++i) { b[i] = lo[i]; b[i + 8] = up[i]; }
    return b;
}

// ---------------------------------------------------------------------------
// Kernel 0: convert 4 weight matrices (512x512 f32, row-major [k][n]) to
// transposed f16 [n][k] so B-fragments are two contiguous 8-half loads.
// ---------------------------------------------------------------------------
__global__ __launch_bounds__(256) void convert_wt(
    const float* __restrict__ fx, const float* __restrict__ fh,
    const float* __restrict__ hf, const float* __restrict__ hx,
    _Float16* __restrict__ fxT, _Float16* __restrict__ fhT,
    _Float16* __restrict__ hfT, _Float16* __restrict__ hxT) {
    int idx = blockIdx.x * blockDim.x + threadIdx.x;   // 0 .. 512*512-1
    int k = idx >> 9;
    int n = idx & 511;
    int t = n * DD + k;
    fxT[t] = (_Float16)fx[idx];
    fhT[t] = (_Float16)fh[idx];
    hfT[t] = (_Float16)hf[idx];
    hxT[t] = (_Float16)hx[idx];
}

// ---------------------------------------------------------------------------
// Kernel 1: parallel precompute over all T*B rows.
//   A = x @ fx        (f16 out, elementwise gate for the recurrent pass)
//   C = x @ hx + bias (f32 out)
// One wave per 64x16 output strip: B-fragments (both weights) loaded once per
// K-step and reused across 4 M-tiles; x A-fragment shared between both WMMAs.
// ---------------------------------------------------------------------------
__global__ __launch_bounds__(256) void mrnn_pre(
    const float* __restrict__ x,          // (T*B, D) f32
    const _Float16* __restrict__ fxT,     // (D, D) f16 transposed
    const _Float16* __restrict__ hxT,     // (D, D) f16 transposed
    const float* __restrict__ bias,       // (D,)
    _Float16* __restrict__ A,             // (T*B, D) f16
    float* __restrict__ C) {              // (T*B, D) f32
    const int wave = (blockIdx.x * blockDim.x + threadIdx.x) >> 5;  // 0..16383
    const int lane = threadIdx.x & 31;
    const int m0 = (wave >> 5) * 64;      // 512 strips of 64 rows
    const int n0 = (wave & 31) * 16;      // 32 column tiles

    v8f accA[4] = {};
    v8f accC[4] = {};
    const _Float16* wfx = fxT + (size_t)n0 * DD;
    const _Float16* whx = hxT + (size_t)n0 * DD;

    for (int k0 = 0; k0 < DD; k0 += 32) {
        v16h b0 = load_bfrag(wfx + k0, DD, lane);
        v16h b1 = load_bfrag(whx + k0, DD, lane);
#pragma unroll
        for (int j = 0; j < 4; ++j) {
            v16h a = load_afrag_f32(x + (size_t)(m0 + 16 * j) * DD + k0, DD, lane);
            accA[j] = __builtin_amdgcn_wmma_f32_16x16x32_f16(false, a, false, b0,
                                                             (short)0, accA[j], false, false);
            accC[j] = __builtin_amdgcn_wmma_f32_16x16x32_f16(false, a, false, b1,
                                                             (short)0, accC[j], false, false);
        }
    }

    const int row_lo = (lane >> 4) * 8;   // C/D layout: M = r + 8*hi
    const int col    = lane & 15;
    const float bv = bias[n0 + col];
#pragma unroll
    for (int j = 0; j < 4; ++j) {
#pragma unroll
        for (int r = 0; r < 8; ++r) {
            const size_t m = (size_t)(m0 + 16 * j + row_lo + r);
            A[m * DD + n0 + col] = (_Float16)accA[j][r];
            C[m * DD + n0 + col] = accC[j][r] + bv;
        }
    }
}

// ---------------------------------------------------------------------------
// Kernel 2: sequential scan. Single persistent 1024-thread workgroup
// (32 waves on one WGP). LDS: h (64K) + ft (64K) + double-buffered A_t stage
// (2x64K) = 256KB of the 320KB WGP LDS.
// Per step:
//   issue async A_{t+1} -> LDS, prefetch C_{t+1}
//   phase 1: G = h @ fh ; ft = A_t * G   (A_t from LDS stage, ft -> LDS)
//   phase 2: h = tanh(ft @ hf + C_t)     (h -> LDS + output)
// Each wave owns one 16-column tile for the whole GEMM: B-fragment loaded
// once per K-step, reused across the 4 M-tiles (4 accumulators).
// ---------------------------------------------------------------------------
__global__ __launch_bounds__(1024) void mrnn_scan(
    const _Float16* __restrict__ A,       // (T*B, D) f16
    const float* __restrict__ C,          // (T*B, D) f32
    const _Float16* __restrict__ fhT,     // (D, D) f16 transposed
    const _Float16* __restrict__ hfT,     // (D, D) f16 transposed
    const float* __restrict__ h0,         // (B, D) f32
    float* __restrict__ hlast,            // (B, D) f32
    float* __restrict__ out) {            // (T, B, D) f32
    __shared__ _Float16 hS[BB * DD];          // 64 KB
    __shared__ _Float16 ftS[BB * DD];         // 64 KB
    __shared__ _Float16 AtS[2][BB * DD];      // 128 KB (double-buffered stage)

    const int tid  = threadIdx.x;
    const int wave = tid >> 5;
    const int lane = tid & 31;
    const int row_lo = (lane >> 4) * 8;
    const int col    = lane & 15;
    const int n0     = wave * 16;         // this wave's column tile (fixed)

    // preload A_0 into stage 0 via the async path; init h from h0
    for (int i = tid; i < (BB * DD) / 8; i += blockDim.x)
        async_copy_b128((unsigned)(uintptr_t)&AtS[0][i * 8], A + i * 8);
    for (int i = tid; i < BB * DD; i += blockDim.x) hS[i] = (_Float16)h0[i];
    wait_asynccnt0();
    __syncthreads();

    for (int t = 0; t < TT; ++t) {
        const int cur = t & 1, nxt = cur ^ 1;
        const float* Ct   = C + (size_t)t * BB * DD;
        float*       outt = out + (size_t)t * BB * DD;

        // stage A_{t+1} (async -> LDS) and warm L2 with C_{t+1}
        if (t + 1 < TT) {
            const _Float16* An = A + (size_t)(t + 1) * BB * DD;
            for (int i = tid; i < (BB * DD) / 8; i += blockDim.x)
                async_copy_b128((unsigned)(uintptr_t)&AtS[nxt][i * 8], An + i * 8);
            const float* Cn = C + (size_t)(t + 1) * BB * DD;
            for (int i = tid; i < (BB * DD) / 16; i += blockDim.x)   // 64B lines
                __builtin_prefetch(Cn + i * 16, 0, 1);
        }

        // ---- phase 1: G = h @ fh ; ft = A_t * G --------------------------
        {
            v8f acc[4] = {};
            const _Float16* wb = fhT + (size_t)n0 * DD;
            for (int k0 = 0; k0 < DD; k0 += 32) {
                v16h b = load_bfrag(wb + k0, DD, lane);
#pragma unroll
                for (int j = 0; j < 4; ++j) {
                    v16h a = load_afrag_f16(&hS[(16 * j) * DD + k0], DD, lane);
                    acc[j] = __builtin_amdgcn_wmma_f32_16x16x32_f16(
                        false, a, false, b, (short)0, acc[j], false, false);
                }
            }
#pragma unroll
            for (int j = 0; j < 4; ++j) {
#pragma unroll
                for (int r = 0; r < 8; ++r) {
                    const int m = 16 * j + row_lo + r;
                    const int n = n0 + col;
                    const float ft = (float)AtS[cur][m * DD + n] * acc[j][r];
                    ftS[m * DD + n] = (_Float16)ft;
                }
            }
        }
        __syncthreads();

        // ---- phase 2: h = tanh(ft @ hf + C_t) ----------------------------
        {
            v8f acc[4] = {};
            const _Float16* wb = hfT + (size_t)n0 * DD;
            for (int k0 = 0; k0 < DD; k0 += 32) {
                v16h b = load_bfrag(wb + k0, DD, lane);
#pragma unroll
                for (int j = 0; j < 4; ++j) {
                    v16h a = load_afrag_f16(&ftS[(16 * j) * DD + k0], DD, lane);
                    acc[j] = __builtin_amdgcn_wmma_f32_16x16x32_f16(
                        false, a, false, b, (short)0, acc[j], false, false);
                }
            }
#pragma unroll
            for (int j = 0; j < 4; ++j) {
#pragma unroll
                for (int r = 0; r < 8; ++r) {
                    const int m = 16 * j + row_lo + r;
                    const int n = n0 + col;
                    const float hn = tanhf(acc[j][r] + Ct[m * DD + n]);
                    outt[m * DD + n] = hn;
                    hS[m * DD + n]   = (_Float16)hn;
                    if (t == TT - 1) hlast[m * DD + n] = hn;
                }
            }
        }
        wait_asynccnt0();   // A_{t+1} stage complete before next-step barrier
        __syncthreads();
    }
}

// ---------------------------------------------------------------------------
// Launcher. Inputs (setup_inputs order): x, h0, fx, fh, hf, hx, bias.
// d_out = [h_last (B*D f32)] ++ [output (T*B*D f32)].
// Workspace layout (bytes):
//   fxT 0 | hxT 512K | fhT 1M | hfT 1.5M | A (f16) 2M..34M | C (f32) 34M..98M
// ---------------------------------------------------------------------------
extern "C" void kernel_launch(void* const* d_in, const int* in_sizes, int n_in,
                              void* d_out, int out_size, void* d_ws, size_t ws_size,
                              hipStream_t stream) {
    const float* x    = (const float*)d_in[0];
    const float* h0   = (const float*)d_in[1];
    const float* fx   = (const float*)d_in[2];
    const float* fh   = (const float*)d_in[3];
    const float* hf   = (const float*)d_in[4];
    const float* hx   = (const float*)d_in[5];
    const float* bias = (const float*)d_in[6];

    char* ws = (char*)d_ws;
    _Float16* fxT  = (_Float16*)(ws + 0);
    _Float16* hxT  = (_Float16*)(ws + (size_t)512 * 1024);
    _Float16* fhT  = (_Float16*)(ws + (size_t)1024 * 1024);
    _Float16* hfT  = (_Float16*)(ws + (size_t)1536 * 1024);
    _Float16* Abuf = (_Float16*)(ws + (size_t)2048 * 1024);
    float*    Cbuf = (float*)   (ws + (size_t)2048 * 1024 +
                                 (size_t)TT * BB * DD * sizeof(_Float16));

    float* hlast = (float*)d_out;
    float* out   = (float*)d_out + (size_t)BB * DD;

    convert_wt<<<(DD * DD) / 256, 256, 0, stream>>>(fx, fh, hf, hx,
                                                    fxT, fhT, hfT, hxT);

    // 512 strips x 32 column tiles = 16384 waves -> 2048 blocks x 256
    mrnn_pre<<<2048, 256, 0, stream>>>(x, fxT, hxT, bias, Abuf, Cbuf);

    // single persistent workgroup for the recurrent scan
    mrnn_scan<<<1, 1024, 0, stream>>>(Abuf, Cbuf, fhT, hfT, h0, hlast, out);
}
